// ScaledDotProductAttention_7413113553036
// MI455X (gfx1250) — matrix operand order repaired and checked by reading, not compiled, not run on hardware
//
#include <hip/hip_runtime.h>

typedef _Float16 half_t;
typedef __attribute__((ext_vector_type(16))) _Float16 v16h;
typedef __attribute__((ext_vector_type(8)))  _Float16 v8h;
typedef __attribute__((ext_vector_type(8)))  float    v8f;

#define S_LEN  4096
#define D_DIM  64
#define N_BH   24            // B*H
#define BM     256           // query rows per workgroup (8 waves x 2 tiles x 16)
#define BN     64            // keys per inner iteration
#define NITER  (S_LEN / BN)  // 64
#define MBLKS  (S_LEN / BM)  // 16
#define QT     2             // query tiles per wave

#define WMMA_F16(A, B, C) \
    __builtin_amdgcn_wmma_f32_16x16x32_f16(false, (A), false, (B), (short)0, (C), false, false)

__launch_bounds__(256, 1)
__global__ void fa_fwd_f16wmma(const float* __restrict__ Qg,
                               const float* __restrict__ Kg,
                               const float* __restrict__ Vg,
                               float* __restrict__ Og)
{
    // Double-buffered K (row-major key,d) and V^T (d,key): 4 * 8KB = 32KB LDS.
    __shared__ __align__(32) half_t ldsK [2][BN * D_DIM];
    __shared__ __align__(32) half_t ldsVt[2][D_DIM * BN];

    const int tid     = threadIdx.x;
    const int lane    = tid & 31;
    const int wave    = tid >> 5;
    const int half_id = lane >> 4;     // which 16-lane half of the wave
    const int l15     = lane & 15;
    const int kb8     = half_id * 8;   // A-matrix interleaved K base (ISA 7.12.2)
    const int kb16    = half_id * 16;  // contiguous K base (unused)

    const int bh    = blockIdx.x / MBLKS;
    const int mblk  = blockIdx.x % MBLKS;
    const int mbase = mblk * BM + wave * 16;   // tile t rows: mbase + t*128

    const size_t head_off = (size_t)bh * S_LEN * D_DIM;
    const float* Qh = Qg + head_off;
    const float* Kh = Kg + head_off;
    const float* Vh = Vg + head_off;
    float*       Oh = Og + head_off;

    // ---- Q tiles as B-operand (Q^T): lane holds query q=l15, d in kb8 table order ----
    // Pre-scaled by log2(e)/sqrt(D) so softmax uses raw exp2.
    v16h qb[QT][2];
    {
        const float scl = 0.125f * 1.44269504088896340736f;
        #pragma unroll
        for (int t = 0; t < QT; ++t) {
            const float* qrow = Qh + (size_t)(mbase + t * 128 + l15) * D_DIM;
            #pragma unroll
            for (int j = 0; j < 16; ++j) {
                qb[t][0][j] = (half_t)(qrow[kb8 + j + (j & 8)]      * scl);
                qb[t][1][j] = (half_t)(qrow[32 + kb8 + j + (j & 8)] * scl);
            }
        }
    }

    // ---- per-lane softmax state for query q = l15 of each tile ----
    float m_run[QT], l_run[QT];
    v8f Oacc[QT][4];
    {
        v8f zero = {};
        #pragma unroll
        for (int t = 0; t < QT; ++t) {
            m_run[t] = -INFINITY; l_run[t] = 0.0f;
            #pragma unroll
            for (int dt = 0; dt < 4; ++dt) Oacc[t][dt] = zero;
        }
    }

    // staging assignments
    const int krow_s = tid >> 2;          // 0..63 : key row this thread stages (K)
    const int kcol_s = (tid & 3) * 16;    // 0/16/32/48 : d base (K)
    const int vd_s   = tid & 63;          // 0..63 : d column this thread stages (V)
    const int vk_s   = (tid >> 6) * 16;   // 0/16/32/48 : key base (V)

    float kreg[16], vreg[16];

    // ---- prologue: load + stage block 0 into buffer 0 ----
    {
        const float* kr = Kh + (size_t)krow_s * D_DIM + kcol_s;
        #pragma unroll
        for (int j = 0; j < 16; ++j) kreg[j] = kr[j];
        #pragma unroll
        for (int j = 0; j < 16; ++j) vreg[j] = Vh[(size_t)(vk_s + j) * D_DIM + vd_s];
        v8h pk0, pk1, pv0, pv1;
        #pragma unroll
        for (int j = 0; j < 8; ++j) {
            pk0[j] = (half_t)kreg[j];  pk1[j] = (half_t)kreg[8 + j];
            pv0[j] = (half_t)vreg[j];  pv1[j] = (half_t)vreg[8 + j];
        }
        *(v8h*)(&ldsK [0][krow_s * D_DIM + kcol_s])     = pk0;
        *(v8h*)(&ldsK [0][krow_s * D_DIM + kcol_s + 8]) = pk1;
        *(v8h*)(&ldsVt[0][vd_s * BN + vk_s])            = pv0;
        *(v8h*)(&ldsVt[0][vd_s * BN + vk_s + 8])        = pv1;
    }

    for (int it = 0; it < NITER; ++it) {
        const int buf = it & 1;
        __syncthreads();

        // ---- issue next block's global loads (overlap with WMMA below) ----
        const bool have_next = (it + 1 < NITER);
        if (have_next) {
            const int nb = (it + 1) * BN;
            const float* kr = Kh + (size_t)(nb + krow_s) * D_DIM + kcol_s;
            #pragma unroll
            for (int j = 0; j < 16; ++j) kreg[j] = kr[j];
            #pragma unroll
            for (int j = 0; j < 16; ++j) vreg[j] = Vh[(size_t)(nb + vk_s + j) * D_DIM + vd_s];
        }
        if (it + 2 < NITER) {   // lowers to global_prefetch_b8
            __builtin_prefetch(Kh + (size_t)((it + 2) * BN + krow_s) * D_DIM + kcol_s, 0, 1);
            __builtin_prefetch(Vh + (size_t)((it + 2) * BN + vk_s) * D_DIM + vd_s, 0, 1);
        }

        // ---- S^T = K * Q^T for both query tiles (K A-operand loaded once) ----
        //      C-layout: lane holds query q=l15; subtile nt rows = keys nt*16+8*half+r.
        v8f sc[QT][4];
        #pragma unroll
        for (int nt = 0; nt < 4; ++nt) {
            const half_t* krw = &ldsK[buf][(nt * 16 + l15) * D_DIM];
            v8h lo0 = *(const v8h*)(krw + kb8);
            v8h hi0 = *(const v8h*)(krw + kb8 + 16);
            v8h lo1 = *(const v8h*)(krw + 32 + kb8);
            v8h hi1 = *(const v8h*)(krw + 32 + kb8 + 16);
            v16h ka0 = __builtin_shufflevector(lo0, hi0, 0,1,2,3,4,5,6,7,8,9,10,11,12,13,14,15);
            v16h ka1 = __builtin_shufflevector(lo1, hi1, 0,1,2,3,4,5,6,7,8,9,10,11,12,13,14,15);
            #pragma unroll
            for (int t = 0; t < QT; ++t) {
                v8f z = {};
                z = WMMA_F16(ka0, qb[t][0], z);
                z = WMMA_F16(ka1, qb[t][1], z);
                sc[t][nt] = z;
            }
        }

        // ---- online softmax (log2 domain) + P pack + O rescale + P*V ----
        v16h pa[QT][2];
        float alpha[QT];
        #pragma unroll
        for (int t = 0; t < QT; ++t) {
            v8f mv = __builtin_elementwise_max(__builtin_elementwise_max(sc[t][0], sc[t][1]),
                                               __builtin_elementwise_max(sc[t][2], sc[t][3]));
            float mloc = fmaxf(fmaxf(fmaxf(mv[0], mv[1]), fmaxf(mv[2], mv[3])),
                               fmaxf(fmaxf(mv[4], mv[5]), fmaxf(mv[6], mv[7])));
            mloc = fmaxf(mloc, __shfl_xor(mloc, 16, 32));   // combine the two halves

            const float mnew = fmaxf(m_run[t], mloc);
            alpha[t] = __builtin_amdgcn_exp2f(m_run[t] - mnew);
            m_run[t] = mnew;

            #pragma unroll
            for (int nt = 0; nt < 4; ++nt)
                #pragma unroll
                for (int r = 0; r < 8; ++r)
                    sc[t][nt][r] = __builtin_amdgcn_exp2f(sc[t][nt][r] - mnew);

            v8f sv = (sc[t][0] + sc[t][1]) + (sc[t][2] + sc[t][3]);
            float ssum = ((sv[0] + sv[1]) + (sv[2] + sv[3])) + ((sv[4] + sv[5]) + (sv[6] + sv[7]));
            ssum += __shfl_xor(ssum, 16, 32);
            l_run[t] = l_run[t] * alpha[t] + ssum;

            // C-layout of S^T == A-layout of P: feed second WMMA directly.
            #pragma unroll
            for (int j = 0; j < 8; ++j) {
                pa[t][0][j]     = (half_t)sc[t][0][j];
                pa[t][0][8 + j] = (half_t)sc[t][1][j];
                pa[t][1][j]     = (half_t)sc[t][2][j];
                pa[t][1][8 + j] = (half_t)sc[t][3][j];
            }
            // rescale O by alpha (broadcast from query-lane to row axis)
            #pragma unroll
            for (int r = 0; r < 8; ++r) {
                const float ar = __shfl(alpha[t], half_id * 8 + r, 32);
                #pragma unroll
                for (int dt = 0; dt < 4; ++dt) Oacc[t][dt][r] *= ar;
            }
        }

        // ---- O += P * V (V B-operand loaded once, used by both tiles) ----
        #pragma unroll
        for (int dt = 0; dt < 4; ++dt) {
            const half_t* vtr = &ldsVt[buf][(dt * 16 + l15) * BN];
            v8h b0l = *(const v8h*)(vtr + kb8);
            v8h b0h = *(const v8h*)(vtr + kb8 + 16);
            v8h b1l = *(const v8h*)(vtr + 32 + kb8);
            v8h b1h = *(const v8h*)(vtr + 32 + kb8 + 16);
            v16h b0 = __builtin_shufflevector(b0l, b0h, 0,1,2,3,4,5,6,7,8,9,10,11,12,13,14,15);
            v16h b1 = __builtin_shufflevector(b1l, b1h, 0,1,2,3,4,5,6,7,8,9,10,11,12,13,14,15);
            #pragma unroll
            for (int t = 0; t < QT; ++t) {
                Oacc[t][dt] = WMMA_F16(pa[t][0], b0, Oacc[t][dt]);
                Oacc[t][dt] = WMMA_F16(pa[t][1], b1, Oacc[t][dt]);
            }
        }

        // ---- convert + store next block into the other LDS buffer ----
        if (have_next) {
            v8h pk0, pk1, pv0, pv1;
            #pragma unroll
            for (int j = 0; j < 8; ++j) {
                pk0[j] = (half_t)kreg[j];  pk1[j] = (half_t)kreg[8 + j];
                pv0[j] = (half_t)vreg[j];  pv1[j] = (half_t)vreg[8 + j];
            }
            *(v8h*)(&ldsK [buf ^ 1][krow_s * D_DIM + kcol_s])     = pk0;
            *(v8h*)(&ldsK [buf ^ 1][krow_s * D_DIM + kcol_s + 8]) = pk1;
            *(v8h*)(&ldsVt[buf ^ 1][vd_s * BN + vk_s])            = pv0;
            *(v8h*)(&ldsVt[buf ^ 1][vd_s * BN + vk_s + 8])        = pv1;
        }
    }

    // ---- epilogue: normalize by row sum, store fp32 ----
    #pragma unroll
    for (int t = 0; t < QT; ++t) {
        const float inv = 1.0f / l_run[t];   // for query q = l15 of tile t
        #pragma unroll
        for (int r = 0; r < 8; ++r) {
            const float ir = __shfl(inv, half_id * 8 + r, 32);
            float* orow = Oh + (size_t)(mbase + t * 128 + half_id * 8 + r) * D_DIM;
            #pragma unroll
            for (int dt = 0; dt < 4; ++dt)
                orow[dt * 16 + l15] = Oacc[t][dt][r] * ir;
        }
    }
}

extern "C" void kernel_launch(void* const* d_in, const int* in_sizes, int n_in,
                              void* d_out, int out_size, void* d_ws, size_t ws_size,
                              hipStream_t stream) {
    (void)in_sizes; (void)n_in; (void)out_size; (void)d_ws; (void)ws_size;
    const float* Q = (const float*)d_in[0];
    const float* K = (const float*)d_in[1];
    const float* V = (const float*)d_in[2];
    float*       O = (float*)d_out;
    dim3 grid(N_BH * MBLKS);   // 24 * 16 = 384 workgroups
    dim3 block(256);           // 8 waves (wave32)
    fa_fwd_f16wmma<<<grid, block, 0, stream>>>(Q, K, V, O);
}
